// TERMAttention_20847771255440
// MI455X (gfx1250) — matrix operand order repaired and checked by reading, not compiled, not run on hardware
//
#include <hip/hip_runtime.h>
#include <hip/hip_bf16.h>

#define AA   32
#define HIDD 256
#define NHH  8

// compiler scheduling fence: stop cross-chain fusion / prefetch hoisting
#define SCHED_FENCE() asm volatile("" ::: "memory")

typedef _Float16 v16h __attribute__((ext_vector_type(16)));
typedef _Float16 v8h  __attribute__((ext_vector_type(8)));
typedef float    v8f  __attribute__((ext_vector_type(8)));

union Frag { v16h v; v8h h[2]; };

// A-matrix fragment (16x32 f16), source row-major [M][K], per ISA 7.12.2:
// lanes 0-15: row=lane, elems 0..7 = K[k0..k0+7], elems 8..15 = K[k0+16..k0+23]
// lanes 16-31: same rows, K offsets +8 / +24
__device__ __forceinline__ v16h load_fragA(const _Float16* base, int row0, int k0,
                                           int ld, int lane) {
  const int r = lane & 15, hf = lane >> 4;
  const _Float16* p = base + (row0 + r) * ld + k0 + hf * 8;
  Frag f;
  f.h[0] = *(const v8h*)(p);
  f.h[1] = *(const v8h*)(p + 16);
  return f.v;
}

// B-matrix fragment (32x16 f16). Storage = B^T row-major (K contiguous per N).
// lanes 0-15: col=lane, elems 0..7 = K[k0..k0+7], elems 8..15 = K[k0+16..k0+23]; lanes 16-31: K offsets +8 / +24
__device__ __forceinline__ v16h load_fragB(const _Float16* base, int col0, int k0,
                                           int ld, int lane) {
  const int c = lane & 15, hf = lane >> 4;
  const _Float16* p = base + (col0 + c) * ld + k0 + hf * 8;
  Frag f;
  f.h[0] = *(const v8h*)(p);
  f.h[1] = *(const v8h*)(p + 16);
  return f.v;
}

__device__ __forceinline__ v8f wmma16(v16h a, v16h b, v8f c) {
  return __builtin_amdgcn_wmma_f32_16x16x32_f16(false, a, false, b, (short)0, c,
                                                false, false);
}

// ---- weight f32 -> f16 conversion (once; weights then live in L2) ----
__global__ __launch_bounds__(256) void cvt_weights(const float* __restrict__ a,
                                                   const float* __restrict__ b,
                                                   const float* __restrict__ c,
                                                   const float* __restrict__ d,
                                                   _Float16* __restrict__ o) {
  int i = blockIdx.x * 256 + threadIdx.x;          // 0 .. 4*65536-1
  const float* srcs[4] = {a, b, c, d};
  o[i] = (_Float16)srcs[i >> 16][i & 65535];
}

// ---- fused TERM attention: 1 workgroup per (b,t) token, wave h = head h ----
// 256-VGPR cap (no s_set_vgpr_msb, 2 WGs/WGP); SCHED_FENCE bounds live set so
// the cap is met without spills.
__global__ __launch_bounds__(256, 4) void term_attn(const float* __restrict__ src,
                                                    const int* __restrict__ mask,
                                                    const _Float16* __restrict__ w16,
                                                    float* __restrict__ out) {
  __shared__ __align__(16) unsigned char smem[65536];   // exactly 64 KB
  const int tid  = threadIdx.x;
  const int h    = tid >> 5;
  const int lane = tid & 31;
  const int c    = lane & 15;
  const int hf   = lane >> 4;
  const size_t token = blockIdx.x;

  _Float16* Xs = (_Float16*)smem;                       // [32][256] src f16 (later: Os)
  const float* srcT = src + token * (AA * HIDD);
  for (int i = tid; i < AA * HIDD; i += 256) Xs[i] = (_Float16)srcT[i];
  __syncthreads();

  // per-head LDS carving (wave-local, no cross-wave hazards)
  _Float16* Qh  = (_Float16*)(smem + 16384 + h * 4096); // [32][32] f16
  _Float16* Kh  = Qh + 1024;                            // [32][32] f16
  _Float16* Vth = (_Float16*)(smem + 49152 + h * 2048); // [32][32] f16 (V transposed [d][a])
  float*    Sh  = (float*)Qh;                           // logits f32 [32][32], aliases Q+K
  _Float16* Ph  = Qh;                                   // probs f16, row stride 64 (in-place)

  const _Float16* WQ = w16;
  const _Float16* WK = w16 + 65536;
  const _Float16* WV = w16 + 131072;
  const _Float16* WO = w16 + 196608;

  // ---- Phase A: Q/K/V head-block projections (wave h -> cols h*32..h*32+31)
  // Hoist the 8 X A-fragments per m-tile; reuse across {Q,K,V} x {n0} (6x).
  for (int m0 = 0; m0 < 32; m0 += 16) {
    v16h xa[8];
#pragma unroll
    for (int kk = 0; kk < 8; ++kk) xa[kk] = load_fragA(Xs, m0, kk * 32, HIDD, lane);

    // -- Q (scaled by 1/sqrt(D))
#pragma unroll
    for (int n0 = 0; n0 < 32; n0 += 16) {
      v8f acc = {};
#pragma unroll
      for (int kk = 0; kk < 8; ++kk)
        acc = wmma16(xa[kk], load_fragB(WQ, h * 32 + n0, kk * 32, HIDD, lane), acc);
#pragma unroll
      for (int v = 0; v < 8; ++v)
        Qh[(m0 + v + 8 * hf) * 32 + n0 + c] = (_Float16)(acc[v] * 0.1767766953f);
      SCHED_FENCE();
    }
    // -- K
#pragma unroll
    for (int n0 = 0; n0 < 32; n0 += 16) {
      v8f acc = {};
#pragma unroll
      for (int kk = 0; kk < 8; ++kk)
        acc = wmma16(xa[kk], load_fragB(WK, h * 32 + n0, kk * 32, HIDD, lane), acc);
#pragma unroll
      for (int v = 0; v < 8; ++v)
        Kh[(m0 + v + 8 * hf) * 32 + n0 + c] = (_Float16)acc[v];
      SCHED_FENCE();
    }
    // -- V (stored transposed: Vt[d][a])
#pragma unroll
    for (int n0 = 0; n0 < 32; n0 += 16) {
      v8f acc = {};
#pragma unroll
      for (int kk = 0; kk < 8; ++kk)
        acc = wmma16(xa[kk], load_fragB(WV, h * 32 + n0, kk * 32, HIDD, lane), acc);
#pragma unroll
      for (int v = 0; v < 8; ++v)
        Vth[(n0 + c) * 32 + (m0 + v + 8 * hf)] = (_Float16)acc[v];
      SCHED_FENCE();
    }
  }

  // ---- Phase B: S = Q * K^T  (all 4 tiles in registers before overwriting Q/K)
  v8f s00 = {}, s01 = {}, s10 = {}, s11 = {};
  {
    v16h q0 = load_fragA(Qh, 0, 0, 32, lane);
    v16h q1 = load_fragA(Qh, 16, 0, 32, lane);
    v16h k0 = load_fragB(Kh, 0, 0, 32, lane);   // B cols of K^T = rows of K
    v16h k1 = load_fragB(Kh, 16, 0, 32, lane);
    s00 = wmma16(q0, k0, s00);
    s01 = wmma16(q0, k1, s01);
    s10 = wmma16(q1, k0, s10);
    s11 = wmma16(q1, k1, s11);
  }
#pragma unroll
  for (int v = 0; v < 8; ++v) {
    const int rl = v + 8 * hf, rh = 16 + v + 8 * hf;
    Sh[rl * 32 + c]      = s00[v];
    Sh[rl * 32 + 16 + c] = s01[v];
    Sh[rh * 32 + c]      = s10[v];
    Sh[rh * 32 + 16 + c] = s11[v];
  }

  // ---- Phase C: masked softmax, lane = row; mask bits from uniform (scalar) loads
  unsigned mb = 0;
  {
    const int* mrow = mask + token * AA;
#pragma unroll
    for (int j = 0; j < 32; ++j) mb |= (mrow[j] > 0 ? 1u : 0u) << j;
  }
  {
    const int r = lane;
    float sv[32];
#pragma unroll
    for (int j = 0; j < 32; ++j) sv[j] = Sh[r * 32 + j];
    float mx = -3.402823466e38f;
#pragma unroll
    for (int j = 0; j < 32; ++j)
      if ((mb >> j) & 1u) mx = fmaxf(mx, sv[j]);
    float pr[32], sum = 0.f;
#pragma unroll
    for (int j = 0; j < 32; ++j) {
      const float e = ((mb >> j) & 1u) ? __expf(sv[j] - mx) : 0.f;
      pr[j] = e;
      sum += e;
    }
    const float inv = (((mb >> r) & 1u) && sum > 0.f) ? 1.f / sum : 0.f;
#pragma unroll
    for (int j = 0; j < 32; ++j)
      Ph[r * 64 + j] = (_Float16)(pr[j] * inv);   // f16 in place over f32 rows
  }

  __syncthreads();   // all waves past their Xs reads -> safe to overwrite with Os

  // ---- Phase D: O_h = P * V  -> write concat buffer Os (aliases Xs)
  _Float16* Os = (_Float16*)smem;                 // [32][256]
#pragma unroll
  for (int m0 = 0; m0 < 32; m0 += 16) {
    v16h af = load_fragA(Ph, m0, 0, 64, lane);    // probs, row stride 64 f16
#pragma unroll
    for (int n0 = 0; n0 < 32; n0 += 16) {
      v16h bf = load_fragB(Vth, n0, 0, 32, lane); // B cols of V = rows of V^T
      v8f acc = {};
      acc = wmma16(af, bf, acc);
#pragma unroll
      for (int v = 0; v < 8; ++v)
        Os[(m0 + v + 8 * hf) * HIDD + h * 32 + n0 + c] = (_Float16)acc[v];
    }
  }
  __syncthreads();

  // ---- Phase E: out = Os * W_O^T  (wave h -> cols h*32..h*32+31), f32 store
  float* outT = out + token * (AA * HIDD);
  for (int m0 = 0; m0 < 32; m0 += 16) {
    v16h oa[8];
#pragma unroll
    for (int kk = 0; kk < 8; ++kk) oa[kk] = load_fragA(Os, m0, kk * 32, HIDD, lane);
#pragma unroll
    for (int n0 = 0; n0 < 32; n0 += 16) {
      v8f acc = {};
#pragma unroll
      for (int kk = 0; kk < 8; ++kk)
        acc = wmma16(oa[kk], load_fragB(WO, h * 32 + n0, kk * 32, HIDD, lane), acc);
#pragma unroll
      for (int v = 0; v < 8; ++v)
        outT[(m0 + v + 8 * hf) * HIDD + h * 32 + n0 + c] = acc[v];
      SCHED_FENCE();
    }
  }
}

extern "C" void kernel_launch(void* const* d_in, const int* in_sizes, int n_in,
                              void* d_out, int out_size, void* d_ws, size_t ws_size,
                              hipStream_t stream) {
  const float* src  = (const float*)d_in[0];
  const int*   mask = (const int*)d_in[1];
  const float* WQ   = (const float*)d_in[2];
  const float* WK   = (const float*)d_in[3];
  const float* WV   = (const float*)d_in[4];
  const float* WO   = (const float*)d_in[5];
  float*       out  = (float*)d_out;
  _Float16*    w16  = (_Float16*)d_ws;          // 4 * 256*256 f16 = 512 KB

  cvt_weights<<<(4 * 65536) / 256, 256, 0, stream>>>(WQ, WK, WV, WO, w16);

  const int tokens = in_sizes[1] / AA;          // B*T = 8192
  term_attn<<<tokens, 256, 0, stream>>>(src, mask, w16, out);
}
